// CausalWanGameActionTransformerBlock_26087631356528
// MI455X (gfx1250) — compile-verified, matrix operand order checked by reading, not run on hardware
//
#include <hip/hip_runtime.h>
#include <hip/hip_bf16.h>
#include <math.h>

#define Bn 1
#define Fn 3
#define Tn 880
#define Sn (Fn * Tn)      // 2640
#define NHn 12
#define HDn 128
#define SCALEF 0.08838834764831845f  // 1/sqrt(128)

typedef __attribute__((ext_vector_type(16))) __bf16 v16bf;
typedef __attribute__((ext_vector_type(8)))  float  v8f;

union Frag {
  v16bf v;
  unsigned short u[16];
  uint4 q[2];
};

__device__ __forceinline__ unsigned short f2bf(float f) {
  unsigned int x = __float_as_uint(f);
  x = (x + 0x7FFFu + ((x >> 16) & 1u)) >> 16;   // round-to-nearest-even
  return (unsigned short)x;
}

// Permuted channel position: within each 32-chunk store groups [0],[2],[1],[3]
// so each wave-half reads its 16 WMMA elements contiguously.
__device__ __forceinline__ int permpos(int ch) {
  const int g = (ch >> 3) & 3;                       // 8-wide group in chunk
  const int pos = ((g & 1) << 4) | ((g >> 1) << 3);  // 0,16,8,24
  return (ch & ~31) + pos + (ch & 7);
}

// Transposed V layout: Vt[((h*8 + ch/16)*16 + ch%16) * Sn + s]
__device__ __forceinline__ size_t vtidx(int h, int ch, int s) {
  return ((size_t)((h * 8 + (ch >> 4)) * 16 + (ch & 15))) * Sn + s;
}

__device__ __forceinline__ v8f wmma_bf16(v16bf a, v16bf b, v8f c) {
  // (neg_a, A, neg_b, B, c_mod, C, reuse_a, reuse_b)
  return __builtin_amdgcn_wmma_f32_16x16x32_bf16(false, a, false, b, (short)0, c, false, false);
}

// ---------------------------------------------------------------------------
// Kernel 1: P = Kpad @ viewmat, Pinv = inv(P), per frame (F=3, trivial)
// ---------------------------------------------------------------------------
__global__ void proj_setup_kernel(const float* __restrict__ viewmats,
                                  const float* __restrict__ Ks,
                                  float* __restrict__ Pout,
                                  float* __restrict__ Piout) {
  int f = threadIdx.x;
  if (f >= Fn) return;
  float Kp[16];
  #pragma unroll
  for (int i = 0; i < 16; ++i) Kp[i] = 0.f;
  #pragma unroll
  for (int i = 0; i < 3; ++i)
    #pragma unroll
    for (int j = 0; j < 3; ++j) Kp[i*4+j] = Ks[f*9 + i*3 + j];
  Kp[15] = 1.f;
  float m[16];
  #pragma unroll
  for (int i = 0; i < 4; ++i)
    #pragma unroll
    for (int j = 0; j < 4; ++j) {
      float acc = 0.f;
      #pragma unroll
      for (int c = 0; c < 4; ++c) acc += Kp[i*4+c] * viewmats[f*16 + c*4 + j];
      m[i*4+j] = acc;
    }
  float inv[16];
  inv[0]  =  m[5]*m[10]*m[15]-m[5]*m[11]*m[14]-m[9]*m[6]*m[15]+m[9]*m[7]*m[14]+m[13]*m[6]*m[11]-m[13]*m[7]*m[10];
  inv[4]  = -m[4]*m[10]*m[15]+m[4]*m[11]*m[14]+m[8]*m[6]*m[15]-m[8]*m[7]*m[14]-m[12]*m[6]*m[11]+m[12]*m[7]*m[10];
  inv[8]  =  m[4]*m[9]*m[15]-m[4]*m[11]*m[13]-m[8]*m[5]*m[15]+m[8]*m[7]*m[13]+m[12]*m[5]*m[11]-m[12]*m[7]*m[9];
  inv[12] = -m[4]*m[9]*m[14]+m[4]*m[10]*m[13]+m[8]*m[5]*m[14]-m[8]*m[6]*m[13]-m[12]*m[5]*m[10]+m[12]*m[6]*m[9];
  inv[1]  = -m[1]*m[10]*m[15]+m[1]*m[11]*m[14]+m[9]*m[2]*m[15]-m[9]*m[3]*m[14]-m[13]*m[2]*m[11]+m[13]*m[3]*m[10];
  inv[5]  =  m[0]*m[10]*m[15]-m[0]*m[11]*m[14]-m[8]*m[2]*m[15]+m[8]*m[3]*m[14]+m[12]*m[2]*m[11]-m[12]*m[3]*m[10];
  inv[9]  = -m[0]*m[9]*m[15]+m[0]*m[11]*m[13]+m[8]*m[1]*m[15]-m[8]*m[3]*m[13]-m[12]*m[1]*m[11]+m[12]*m[3]*m[9];
  inv[13] =  m[0]*m[9]*m[14]-m[0]*m[10]*m[13]-m[8]*m[1]*m[14]+m[8]*m[2]*m[13]+m[12]*m[1]*m[10]-m[12]*m[2]*m[9];
  inv[2]  =  m[1]*m[6]*m[15]-m[1]*m[7]*m[14]-m[5]*m[2]*m[15]+m[5]*m[3]*m[14]+m[13]*m[2]*m[7]-m[13]*m[3]*m[6];
  inv[6]  = -m[0]*m[6]*m[15]+m[0]*m[7]*m[14]+m[4]*m[2]*m[15]-m[4]*m[3]*m[14]-m[12]*m[2]*m[7]+m[12]*m[3]*m[6];
  inv[10] =  m[0]*m[5]*m[15]-m[0]*m[7]*m[13]-m[4]*m[1]*m[15]+m[4]*m[3]*m[13]+m[12]*m[1]*m[7]-m[12]*m[3]*m[5];
  inv[14] = -m[0]*m[5]*m[14]+m[0]*m[6]*m[13]+m[4]*m[1]*m[14]-m[4]*m[2]*m[13]-m[12]*m[1]*m[6]+m[12]*m[2]*m[5];
  inv[3]  = -m[1]*m[6]*m[11]+m[1]*m[7]*m[10]+m[5]*m[2]*m[11]-m[5]*m[3]*m[10]-m[9]*m[2]*m[7]+m[9]*m[3]*m[6];
  inv[7]  =  m[0]*m[6]*m[11]-m[0]*m[7]*m[10]-m[4]*m[2]*m[11]+m[4]*m[3]*m[10]+m[8]*m[2]*m[7]-m[8]*m[3]*m[6];
  inv[11] = -m[0]*m[5]*m[11]+m[0]*m[7]*m[9]+m[4]*m[1]*m[11]-m[4]*m[3]*m[9]-m[8]*m[1]*m[7]+m[8]*m[3]*m[5];
  inv[15] =  m[0]*m[5]*m[10]-m[0]*m[6]*m[9]-m[4]*m[1]*m[10]+m[4]*m[2]*m[9]+m[8]*m[1]*m[6]-m[8]*m[2]*m[5];
  float det = m[0]*inv[0] + m[1]*inv[4] + m[2]*inv[8] + m[3]*inv[12];
  float rd = 1.0f / det;
  #pragma unroll
  for (int i = 0; i < 16; ++i) { Pout[f*16+i] = m[i]; Piout[f*16+i] = inv[i]*rd; }
}

// ---------------------------------------------------------------------------
// Kernel 2: fused RoPE(q,k) + bf16 convert; q,k -> permuted-channel layout,
// v -> transposed-chunk layout.
// ---------------------------------------------------------------------------
__global__ void rope_convert_kernel(const float* __restrict__ q,
                                    const float* __restrict__ k,
                                    const float* __restrict__ v,
                                    const float* __restrict__ cosb,
                                    const float* __restrict__ sinb,
                                    unsigned short* __restrict__ qr,
                                    unsigned short* __restrict__ kr,
                                    unsigned short* __restrict__ vt) {
  int idx = blockIdx.x * blockDim.x + threadIdx.x;
  const int total = Sn * NHn * (HDn / 2);
  if (idx >= total) return;
  int p = idx % (HDn / 2);
  int t = idx / (HDn / 2);
  int h = t % NHn;
  int s = t / NHn;
  const int ch = 2 * p;
  size_t base = ((size_t)s * NHn + h) * HDn + ch;
  size_t pbase = ((size_t)s * NHn + h) * HDn + permpos(ch);  // pair stays contiguous
  float c  = cosb[s * (HDn/2) + p];
  float sn = sinb[s * (HDn/2) + p];
  float q1 = q[base], q2 = q[base+1];
  qr[pbase]   = f2bf(q1*c - q2*sn);
  qr[pbase+1] = f2bf(q2*c + q1*sn);
  float k1 = k[base], k2 = k[base+1];
  kr[pbase]   = f2bf(k1*c - k2*sn);
  kr[pbase+1] = f2bf(k2*c + k1*sn);
  vt[vtidx(h, ch,     s)] = f2bf(v[base]);
  vt[vtidx(h, ch + 1, s)] = f2bf(v[base+1]);
}

// ---------------------------------------------------------------------------
// Kernel 3: per-frame 4x4 group projections: qp = P*q4 (permuted layout),
// kp = Pinv^T*k4 (permuted layout), vp = Pinv^T*v4 (transposed layout)
// ---------------------------------------------------------------------------
__global__ void project_inputs_kernel(const float* __restrict__ q,
                                      const float* __restrict__ k,
                                      const float* __restrict__ v,
                                      const float* __restrict__ Pm,
                                      const float* __restrict__ Pi,
                                      unsigned short* __restrict__ qp,
                                      unsigned short* __restrict__ kp,
                                      unsigned short* __restrict__ vpt) {
  int idx = blockIdx.x * blockDim.x + threadIdx.x;
  const int total = Sn * NHn * (HDn / 4);
  if (idx >= total) return;
  int g = idx % (HDn / 4);
  int t = idx / (HDn / 4);
  int h = t % NHn;
  int s = t / NHn;
  int f = s / Tn;
  const float* P = Pm + f * 16;
  const float* Q = Pi + f * 16;
  const int ch0 = 4 * g;
  size_t base  = ((size_t)s * NHn + h) * HDn + ch0;
  size_t pbase = ((size_t)s * NHn + h) * HDn + permpos(ch0);  // 4 channels contiguous
  float xq[4], xk[4], xv[4];
  #pragma unroll
  for (int c = 0; c < 4; ++c) { xq[c] = q[base+c]; xk[c] = k[base+c]; xv[c] = v[base+c]; }
  #pragma unroll
  for (int d = 0; d < 4; ++d) {
    float aq = 0.f, ak = 0.f, av = 0.f;
    #pragma unroll
    for (int c = 0; c < 4; ++c) {
      aq += P[d*4+c] * xq[c];        // 'bfdc' : P[d][c]
      ak += Q[c*4+d] * xk[c];        // 'bfcd' : Pinv[c][d]
      av += Q[c*4+d] * xv[c];
    }
    qp[pbase + d] = f2bf(aq);
    kp[pbase + d] = f2bf(ak);
    vpt[vtidx(h, ch0 + d, s)] = f2bf(av);
  }
}

// ---------------------------------------------------------------------------
// Kernel 4: flash attention, one wave per 16-query tile per head.
// Scores computed transposed (S^T = K * Q^T) so the score C-fragments map
// directly onto the A-fragment of P for the PV WMMA (no LDS transpose).
// All fragment loads are contiguous b128 pairs via the staged layouts.
// ---------------------------------------------------------------------------
__global__ __launch_bounds__(32)
void flash_attn_kernel(const unsigned short* __restrict__ Qb,
                       const unsigned short* __restrict__ Kb,
                       const unsigned short* __restrict__ Vt,
                       float* __restrict__ Out) {
  const int lane = threadIdx.x;
  const int hf = lane >> 4;     // half of wave (0/1)
  const int ln = lane & 15;
  const int q0 = blockIdx.x * 16;
  const int h  = blockIdx.y;
  const int limit = ((q0 / Tn) + 1) * Tn;   // exclusive key bound (frame causal)

  // Q as B-fragments (lane = query, elements = channels), 4 chunks of K=32
  Frag Bq[4];
  {
    const unsigned short* qrow = Qb + ((size_t)(q0 + ln) * NHn + h) * HDn;
    #pragma unroll
    for (int c = 0; c < 4; ++c) {
      const unsigned short* p = qrow + c * 32 + 16 * hf;
      Bq[c].q[0] = *(const uint4*)(p);
      Bq[c].q[1] = *(const uint4*)(p + 8);
    }
  }

  v8f O[8];
  #pragma unroll
  for (int i = 0; i < 8; ++i) O[i] = (v8f){};
  float mrun = -3.0e30f, lrun = 0.f;

  for (int kb = 0; kb < limit; kb += 32) {
    const bool has2 = (kb + 16) < limit;
    const int kb2 = has2 ? (kb + 16) : kb;   // masked half: reload valid finite data

    // ---- scores S^T for 32-key block (two 16x16 fragments) ----
    v8f s0 = (v8f){}, s1 = (v8f){};
    {
      const unsigned short* krow = Kb + ((size_t)(kb + ln) * NHn + h) * HDn;
      #pragma unroll
      for (int c = 0; c < 4; ++c) {
        Frag aK;
        const unsigned short* p = krow + c * 32 + 16 * hf;
        aK.q[0] = *(const uint4*)(p);
        aK.q[1] = *(const uint4*)(p + 8);
        s0 = wmma_bf16(aK.v, Bq[c].v, s0);
      }
    }
    if (has2) {
      const unsigned short* krow = Kb + ((size_t)(kb + 16 + ln) * NHn + h) * HDn;
      #pragma unroll
      for (int c = 0; c < 4; ++c) {
        Frag aK;
        const unsigned short* p = krow + c * 32 + 16 * hf;
        aK.q[0] = *(const uint4*)(p);
        aK.q[1] = *(const uint4*)(p + 8);
        s1 = wmma_bf16(aK.v, Bq[c].v, s1);
      }
    }

    // ---- online softmax (lane = query; single cross-half reduction) ----
    float mloc = -3.0e30f;
    #pragma unroll
    for (int r = 0; r < 8; ++r) mloc = fmaxf(mloc, s0[r]);
    if (has2) {
      #pragma unroll
      for (int r = 0; r < 8; ++r) mloc = fmaxf(mloc, s1[r]);
    }
    mloc *= SCALEF;
    mloc = fmaxf(mloc, __shfl_xor(mloc, 16, 32));
    const float mnew  = fmaxf(mrun, mloc);
    const float alpha = __expf(mrun - mnew);
    float p0[8], p1[8];
    float rs = 0.f;
    #pragma unroll
    for (int r = 0; r < 8; ++r) { p0[r] = __expf(s0[r]*SCALEF - mnew); rs += p0[r]; }
    #pragma unroll
    for (int r = 0; r < 8; ++r) { p1[r] = has2 ? __expf(s1[r]*SCALEF - mnew) : 0.f; rs += p1[r]; }
    rs += __shfl_xor(rs, 16, 32);
    lrun = lrun * alpha + rs;
    mrun = mnew;

    // per-output-row alpha (row = r + 8*hf in O fragments)
    float arow[8];
    #pragma unroll
    for (int r = 0; r < 8; ++r) arow[r] = __shfl(alpha, r + 8*hf, 32);
    #pragma unroll
    for (int c8 = 0; c8 < 8; ++c8) {
      #pragma unroll
      for (int r = 0; r < 8; ++r) O[c8][r] *= arow[r];
    }

    // ---- P fragment: direct repack of score fragments (no LDS) ----
    Frag pf;
    #pragma unroll
    for (int e = 0; e < 8; ++e) pf.u[e] = f2bf(p0[e]);
    #pragma unroll
    for (int e = 0; e < 8; ++e) pf.u[8+e] = f2bf(p1[e]);

    // ---- O += P * V (8 WMMAs over HD=128); V fragment = 2 b128 loads ----
    #pragma unroll
    for (int c8 = 0; c8 < 8; ++c8) {
      Frag vf;
      const unsigned short* vcol = Vt + ((size_t)((h * 8 + c8) * 16 + ln)) * Sn;
      vf.q[0] = *(const uint4*)(vcol + kb  + 8 * hf);
      vf.q[1] = *(const uint4*)(vcol + kb2 + 8 * hf);
      O[c8] = wmma_bf16(pf.v, vf.v, O[c8]);
    }
  }

  // ---- epilogue: divide by row sums, store f32 ----
  const float linv = 1.0f / lrun;
  float lrow[8];
  #pragma unroll
  for (int r = 0; r < 8; ++r) lrow[r] = __shfl(linv, r + 8*hf, 32);
  #pragma unroll
  for (int c8 = 0; c8 < 8; ++c8) {
    #pragma unroll
    for (int r = 0; r < 8; ++r) {
      const int row = q0 + r + 8*hf;
      Out[((size_t)row * NHn + h) * HDn + c8*16 + ln] = O[c8][r] * lrow[r];
    }
  }
}

// ---------------------------------------------------------------------------
// Kernel 5: prope_out = P * po (4-groups), f32 -> f32
// ---------------------------------------------------------------------------
__global__ void project_out_kernel(const float* __restrict__ po,
                                   const float* __restrict__ Pm,
                                   float* __restrict__ out) {
  int idx = blockIdx.x * blockDim.x + threadIdx.x;
  const int total = Sn * NHn * (HDn / 4);
  if (idx >= total) return;
  int g = idx % (HDn / 4);
  int t = idx / (HDn / 4);
  int h = t % NHn;
  int s = t / NHn;
  int f = s / Tn;
  const float* P = Pm + f * 16;
  size_t base = ((size_t)s * NHn + h) * HDn + 4 * g;
  float x[4];
  #pragma unroll
  for (int c = 0; c < 4; ++c) x[c] = po[base + c];
  #pragma unroll
  for (int d = 0; d < 4; ++d) {
    out[base + d] = P[d*4+0]*x[0] + P[d*4+1]*x[1] + P[d*4+2]*x[2] + P[d*4+3]*x[3];
  }
}

// ---------------------------------------------------------------------------
extern "C" void kernel_launch(void* const* d_in, const int* in_sizes, int n_in,
                              void* d_out, int out_size, void* d_ws, size_t ws_size,
                              hipStream_t stream) {
  (void)in_sizes; (void)n_in; (void)out_size; (void)ws_size;
  const float* q        = (const float*)d_in[0];
  const float* k        = (const float*)d_in[1];
  const float* v        = (const float*)d_in[2];
  const float* cosb     = (const float*)d_in[3];
  const float* sinb     = (const float*)d_in[4];
  const float* viewmats = (const float*)d_in[5];
  const float* Ks       = (const float*)d_in[6];
  float* out = (float*)d_out;

  const size_t N = (size_t)Sn * NHn * HDn;   // 4,055,040 elems
  char* ws = (char*)d_ws;
  size_t off = 0;
  auto take = [&](size_t bytes) -> char* {
    char* p = ws + off;
    off = (off + bytes + 255) & ~(size_t)255;
    return p;
  };
  unsigned short* qr  = (unsigned short*)take(N * 2);
  unsigned short* kr  = (unsigned short*)take(N * 2);
  unsigned short* vt  = (unsigned short*)take(N * 2);  // followed by more ws (safe tail reads)
  unsigned short* qp  = (unsigned short*)take(N * 2);
  unsigned short* kp  = (unsigned short*)take(N * 2);
  unsigned short* vpt = (unsigned short*)take(N * 2);  // followed by po (safe tail reads)
  float* po  = (float*)take(N * 4);
  float* Pw  = (float*)take(Fn * 16 * 4);
  float* Piw = (float*)take(Fn * 16 * 4);

  proj_setup_kernel<<<1, 4, 0, stream>>>(viewmats, Ks, Pw, Piw);

  {
    const int tot = Sn * NHn * (HDn / 2);
    rope_convert_kernel<<<(tot + 255) / 256, 256, 0, stream>>>(q, k, v, cosb, sinb, qr, kr, vt);
  }
  {
    const int tot = Sn * NHn * (HDn / 4);
    project_inputs_kernel<<<(tot + 255) / 256, 256, 0, stream>>>(q, k, v, Pw, Piw, qp, kp, vpt);
  }

  dim3 agrid(Sn / 16, NHn);
  flash_attn_kernel<<<agrid, 32, 0, stream>>>(qr, kr, vt, out);       // rope path -> out[0..N)
  flash_attn_kernel<<<agrid, 32, 0, stream>>>(qp, kp, vpt, po);       // prope path -> ws

  {
    const int tot = Sn * NHn * (HDn / 4);
    project_out_kernel<<<(tot + 255) / 256, 256, 0, stream>>>(po, Pw, out + N);
  }
}
